// LinearInt4_37950331027889
// MI455X (gfx1250) — matrix operand order repaired and checked by reading, not compiled, not run on hardware
//
#include <hip/hip_runtime.h>
#include <hip/hip_bf16.h>
#include <stdint.h>

// ---------------------------------------------------------------------------
// Int4 group-quantized linear: out[t,o] = sum_k x[t,k] * ((q[o,k]-8)*s[g,o] + z[g,o])
// M = 8192 tokens, K = 4096, N = 11008, group = 128.
//
// MI455X plan: compute-bound GEMM (AI ~1100 FLOP/B >> machine balance), so:
//  - pre-pass: X fp32->bf16 (64 MB) and W int4->bf16 dequant (90 MB) into ws.
//    154 MB working set is L2-resident (192 MB L2).
//  - GEMM: pure bf16 WMMA (v_wmma_f32_16x16x32_bf16), tiles streamed with
//    global_load_async_to_lds_b128 (ASYNCcnt), double-buffered XOR-swizzled LDS.
//  - fallback template instantiation with fused dequant if ws too small.
// ---------------------------------------------------------------------------

#define IN_F   4096
#define OUT_F  11008
#define GS     128

#define BM     128
#define BN     128
#define BK     64                 // halves per K-step (two 16x16x32 sub-steps)
#define NKT    (IN_F / BK)        // 64
static_assert((NKT & 1) == 0, "K loop unrolled by 2");

// LDS: 4 buffers of 128 rows x 64 halves (128 B rows, 16B-chunk XOR swizzle).
#define AB1    8192               // half-offsets of the 4 buffers
#define BB0    16384
#define BB1    24576

typedef __attribute__((ext_vector_type(16))) __bf16 bf16x16;
typedef __attribute__((ext_vector_type(8)))  float  f32x8;
union AFrag { bf16x16 v; uint4 q[2]; };
template<int N> struct IC { static constexpr int value = N; };

__device__ __forceinline__ unsigned short f2bf(float f) {
    // round-to-nearest-even fp32 -> bf16 (NaN edge ignored)
    unsigned int u = __builtin_bit_cast(unsigned int, f);
    unsigned int r = u + 0x7FFFu + ((u >> 16) & 1u);
    return (unsigned short)(r >> 16);
}

__device__ __forceinline__ uint2 pack4(float a, float b, float c, float d) {
    return make_uint2((unsigned)f2bf(a) | ((unsigned)f2bf(b) << 16),
                      (unsigned)f2bf(c) | ((unsigned)f2bf(d) << 16));
}

// XOR-swizzled half-index within one 128x64 buffer: 16B chunk c of row r lives
// at chunk (c ^ (r&7)). Keeps every b128 access 16B-aligned with no padding.
__device__ __forceinline__ unsigned swh(int row, int halfoff) {
    const int cb = halfoff >> 3;
    const int in = halfoff & 7;
    return (unsigned)(row * 64 + (((cb ^ (row & 7)) << 3) | in));
}

// ---------------------------------------------------------------------------
// prep kernels
// ---------------------------------------------------------------------------
__global__ __launch_bounds__(256)
void cvt_x_bf16(const float* __restrict__ X, unsigned short* __restrict__ Xb, int n4) {
    const int i = blockIdx.x * 256 + threadIdx.x;   // 4 elements per thread
    if (i >= n4) return;
    const float4 v = ((const float4*)X)[i];
    ((uint2*)Xb)[i] = pack4(v.x, v.y, v.z, v.w);
}

__global__ __launch_bounds__(256)
void deq_w_bf16(const int* __restrict__ Wq, const float* __restrict__ SZ,
                unsigned short* __restrict__ Wb, int n4) {
    const int i = blockIdx.x * 256 + threadIdx.x;   // 4 codes along k per thread
    if (i >= n4) return;
    const int o = i >> 10;                          // K/4 = 1024 chunks per row
    const int g = (i & 1023) >> 5;                  // 32 codes-chunks per group
    const float2 sz = *(const float2*)(SZ + ((size_t)g * OUT_F + o) * 2);
    const float s = sz.x, z8 = sz.y - 8.0f * sz.x;  // (q-8)*s+z = q*s + (z-8s)
    const int4 q = ((const int4*)Wq)[i];
    ((uint2*)Wb)[i] = pack4(fmaf((float)q.x, s, z8), fmaf((float)q.y, s, z8),
                            fmaf((float)q.z, s, z8), fmaf((float)q.w, s, z8));
}

// ---------------------------------------------------------------------------
// GEMM: PRE=true reads prepped bf16 via async-to-LDS; PRE=false fuses dequant.
// ---------------------------------------------------------------------------
template<bool PRE>
__global__ __launch_bounds__(256)
void int4_gemm_wmma(const float* __restrict__ X, const int* __restrict__ Wq,
                    const float* __restrict__ SZ,
                    const unsigned short* __restrict__ Xb,
                    const unsigned short* __restrict__ Wb,
                    float* __restrict__ Out)
{
    __shared__ unsigned short smem[4 * BM * BK];    // 64 KB, 4 buffers

    const int tid   = threadIdx.x;
    const int lane  = tid & 31;
    const int wave  = tid >> 5;
    const int m0    = blockIdx.y * BM;
    const int n0    = blockIdx.x * BN;
    const int wm    = (wave >> 2) * 64;             // 2x4 wave grid, 64x32 tiles
    const int wn    = (wave & 3)  * 32;
    const int lhalf = lane >> 4;
    const int l16   = lane & 15;

    // kt-invariant fragment LDS indices (halves, swizzled)
    unsigned aIdx[2][4][2], bIdx[2][2][2];
    #pragma unroll
    for (int kk = 0; kk < 2; ++kk) {
        #pragma unroll
        for (int t = 0; t < 4; ++t) {               // A: elems 0..7 -> k base..+7, 8..15 -> +16..+23
            const int row = wm + t * 16 + l16;
            const int cb0 = kk * 4 + lhalf;
            aIdx[kk][t][0] = row * 64 + (((cb0    ) ^ (row & 7)) << 3);
            aIdx[kk][t][1] = row * 64 + (((cb0 + 2) ^ (row & 7)) << 3);
        }
        #pragma unroll
        for (int t = 0; t < 2; ++t) {               // B: elems 0..7 -> k base..+7, 8..15 -> +16..+23
            const int row = wn + t * 16 + l16;
            const int cb0 = kk * 4 + lhalf;
            bIdx[kk][t][0] = row * 64 + (((cb0    ) ^ (row & 7)) << 3);
            bIdx[kk][t][1] = row * 64 + (((cb0 + 2) ^ (row & 7)) << 3);
        }
    }

    // async staging state (PRE): one row-half (4 x 16B chunks) per thread.
    // LDS addrs pre-biased by -i*16 so the shared IOFFSET restores them.
    const char* gA = nullptr;
    const char* gB = nullptr;
    unsigned ldsA[2][4], ldsB[2][4];
    if constexpr (PRE) {
        const int row = tid >> 1;
        const int ch0 = (tid & 1) * 4;
        #pragma unroll
        for (int i = 0; i < 4; ++i) {
            const unsigned off = (unsigned)(row * 64 + (((ch0 + i) ^ (row & 7)) << 3));
            ldsA[0][i] = (unsigned)(uintptr_t)(smem + off)       - i * 16;
            ldsA[1][i] = (unsigned)(uintptr_t)(smem + AB1 + off) - i * 16;
            ldsB[0][i] = (unsigned)(uintptr_t)(smem + BB0 + off) - i * 16;
            ldsB[1][i] = (unsigned)(uintptr_t)(smem + BB1 + off) - i * 16;
        }
        gA = (const char*)(Xb + (size_t)(m0 + row) * IN_F) + (tid & 1) * 64;
        gB = (const char*)(Wb + (size_t)(n0 + row) * IN_F) + (tid & 1) * 64;
    }

    f32x8 acc[4][2] = {};

    auto stage = [&](int ktNext, auto bufc) {
        constexpr int BUF = decltype(bufc)::value;
        if constexpr (PRE) {
            // 8 x ASYNCcnt-tracked 16B/lane copies, LDS written by the async unit
            asm volatile(
                "global_load_async_to_lds_b128 %0, %8, off\n\t"
                "global_load_async_to_lds_b128 %1, %8, off offset:16\n\t"
                "global_load_async_to_lds_b128 %2, %8, off offset:32\n\t"
                "global_load_async_to_lds_b128 %3, %8, off offset:48\n\t"
                "global_load_async_to_lds_b128 %4, %9, off\n\t"
                "global_load_async_to_lds_b128 %5, %9, off offset:16\n\t"
                "global_load_async_to_lds_b128 %6, %9, off offset:32\n\t"
                "global_load_async_to_lds_b128 %7, %9, off offset:48"
                :: "v"(ldsA[BUF][0]), "v"(ldsA[BUF][1]), "v"(ldsA[BUF][2]), "v"(ldsA[BUF][3]),
                   "v"(ldsB[BUF][0]), "v"(ldsB[BUF][1]), "v"(ldsB[BUF][2]), "v"(ldsB[BUF][3]),
                   "v"(gA), "v"(gB)
                : "memory");
            gA += BK * 2;       // advance one K-step (128 B)
            gB += BK * 2;
        } else {
            const int lrow = tid >> 4;
            const int lk   = (tid & 15) * 4;
            const int k0   = ktNext * BK;
            const int g    = k0 >> 7;
            const unsigned ab = BUF ? AB1 : 0u;
            const unsigned bb = BUF ? BB1 : (unsigned)BB0;
            #pragma unroll
            for (int p = 0; p < 8; ++p) {           // A: fp32 -> bf16
                const int m = lrow + p * 16;
                const float4 v = *(const float4*)(X + (size_t)(m0 + m) * IN_F + k0 + lk);
                *(uint2*)&smem[ab + swh(m, lk)] = pack4(v.x, v.y, v.z, v.w);
            }
            #pragma unroll
            for (int p = 0; p < 8; ++p) {           // B: int4 codes -> bf16
                const int n = lrow + p * 16;
                const int o = n0 + n;
                const float2 sz = *(const float2*)(SZ + ((size_t)g * OUT_F + o) * 2);
                const float s = sz.x, z8 = sz.y - 8.0f * sz.x;
                const int4 q = *(const int4*)(Wq + (size_t)o * IN_F + k0 + lk);
                *(uint2*)&smem[bb + swh(n, lk)] =
                    pack4(fmaf((float)q.x, s, z8), fmaf((float)q.y, s, z8),
                          fmaf((float)q.z, s, z8), fmaf((float)q.w, s, z8));
            }
        }
    };

    auto computeStep = [&](auto curc) {
        constexpr int CUR = decltype(curc)::value;
        const unsigned short* sA = smem + (CUR ? AB1 : 0);
        const unsigned short* sB = smem + (CUR ? BB1 : BB0);
        #pragma unroll
        for (int kk = 0; kk < 2; ++kk) {
            AFrag a[4], b[2];
            #pragma unroll
            for (int t = 0; t < 4; ++t) {
                a[t].q[0] = *(const uint4*)&sA[aIdx[kk][t][0]];
                a[t].q[1] = *(const uint4*)&sA[aIdx[kk][t][1]];
            }
            #pragma unroll
            for (int t = 0; t < 2; ++t) {
                b[t].q[0] = *(const uint4*)&sB[bIdx[kk][t][0]];
                b[t].q[1] = *(const uint4*)&sB[bIdx[kk][t][1]];
            }
            #pragma unroll
            for (int i = 0; i < 4; ++i)
                #pragma unroll
                for (int j = 0; j < 2; ++j)
                    acc[i][j] = __builtin_amdgcn_wmma_f32_16x16x32_bf16(
                        false, a[i].v, false, b[j].v, (short)0, acc[i][j], false, false);
        }
    };

    stage(0, IC<0>{});                              // prologue fill of buffer 0

    for (int kt = 0; kt < NKT; kt += 2) {
        // --- even sub-iteration: compute buf0, prefetch buf1 ---
        if constexpr (PRE) asm volatile("s_wait_asynccnt 0x0" ::: "memory");
        __syncthreads();                            // buf0 visible; buf1 free to overwrite
        stage(kt + 1, IC<1>{});                     // kt+1 < NKT always (NKT even)
        computeStep(IC<0>{});
        // --- odd sub-iteration: compute buf1, prefetch buf0 ---
        if constexpr (PRE) asm volatile("s_wait_asynccnt 0x0" ::: "memory");
        __syncthreads();
        if (kt + 2 < NKT) stage(kt + 2, IC<0>{});
        computeStep(IC<1>{});
    }

    // epilogue: C/D layout -> global (row m = (lane/16)*8 + r, col n = lane%16)
    #pragma unroll
    for (int i = 0; i < 4; ++i) {
        #pragma unroll
        for (int j = 0; j < 2; ++j) {
            const int mbase = m0 + wm + i * 16 + lhalf * 8;
            const int nidx  = n0 + wn + j * 16 + l16;
            #pragma unroll
            for (int r = 0; r < 8; ++r)
                Out[(size_t)(mbase + r) * OUT_F + nidx] = acc[i][j][r];
        }
    }
}

// ---------------------------------------------------------------------------
extern "C" void kernel_launch(void* const* d_in, const int* in_sizes, int n_in,
                              void* d_out, int out_size, void* d_ws, size_t ws_size,
                              hipStream_t stream) {
    const float* X  = (const float*)d_in[0];    // [4, 2048, 4096] fp32
    const int*   Wq = (const int*)  d_in[1];    // [11008, 4096] int32 codes 0..15
    const float* SZ = (const float*)d_in[2];    // [32, 11008, 2] fp32
    float* Out = (float*)d_out;                 // [4, 2048, 11008] fp32

    const int M = in_sizes[0] / IN_F;           // 8192
    const size_t needA = (size_t)M * IN_F * 2;          // X in bf16: 64 MB
    const size_t needB = (size_t)OUT_F * IN_F * 2;      // W in bf16: 90 MB

    dim3 grid(OUT_F / BN, M / BM);              // (86, 64)
    dim3 block(256);                            // 8 wave32 waves

    if (ws_size >= needA + needB) {
        unsigned short* Xb = (unsigned short*)d_ws;
        unsigned short* Wb = (unsigned short*)((char*)d_ws + needA);
        const int nx4 = M * IN_F / 4;
        const int nw4 = OUT_F * IN_F / 4;
        cvt_x_bf16<<<(nx4 + 255) / 256, 256, 0, stream>>>(X, Xb, nx4);
        deq_w_bf16<<<(nw4 + 255) / 256, 256, 0, stream>>>(Wq, SZ, Wb, nw4);
        int4_gemm_wmma<true><<<grid, block, 0, stream>>>(X, Wq, SZ, Xb, Wb, Out);
    } else {
        int4_gemm_wmma<false><<<grid, block, 0, stream>>>(X, Wq, SZ, nullptr, nullptr, Out);
    }
}
